// SingleMambaBlock_8641474199631
// MI455X (gfx1250) — matrix operand order repaired and checked by reading, not compiled, not run on hardware
//
#include <hip/hip_runtime.h>
#include <math.h>

// ---------------- problem constants ----------------
#define BATCH   4
#define LSEQ    2048
#define DMODEL  768
#define DINNER  1536
#define XZDIM   3072            // 2*DINNER
#define NTOK    (BATCH*LSEQ)    // 8192 tokens
#define DTRANK  48
#define DTPAD   64              // DTRANK padded to WMMA K granularity
#define DSTATE  16
#define DBCN    80              // DTRANK + 2*DSTATE
#define DCONV   4

typedef __attribute__((ext_vector_type(16))) __bf16 v16bf;
typedef __attribute__((ext_vector_type(8)))  __bf16 v8bf;
typedef __attribute__((ext_vector_type(8)))  float  v8f;

// ---------------- WMMA fragment helpers (CDNA5 16x16x32 bf16 layouts) ----------------
// A (16x32, MxK): lanes 0-15 hold M=0..15 with K {0..7,16..23}; lanes 16-31 K {8..15,24..31}
__device__ inline v16bf load_A_frag(const __bf16* __restrict__ A, int lda,
                                    int row0, int k0, int lane) {
  int m  = row0 + (lane & 15);
  int kb = (lane >> 4) * 8;
  const __bf16* p = A + (size_t)m * lda + k0 + kb;
  v8bf lo = *(const v8bf*)(p);        // K = kb+0..7      (16B aligned)
  v8bf hi = *(const v8bf*)(p + 16);   // K = kb+16..23
  return __builtin_shufflevector(lo, hi, 0,1,2,3,4,5,6,7,8,9,10,11,12,13,14,15);
}

// B (32x16, KxN) fragment from an LDS-staged weight tile laid out [rows][PADK] bf16.
// lane n = lane&15; lanes 0-15 read K {0..7,16..23}, lanes 16-31 read K {8..15,24..31} (matches A).
#define PADK 40   // 32 K-elements + 8 pad (80B row stride: b128 reads spread across banks)
__device__ inline v16bf load_B_frag_lds(const __bf16* buf, int j, int lane) {
  int n  = j * 16 + (lane & 15);
  int kb = (lane >> 4) * 8;
  const __bf16* p = buf + n * PADK + kb;
  v8bf lo = *(const v8bf*)(p);
  v8bf hi = *(const v8bf*)(p + 16);
  return __builtin_shufflevector(lo, hi, 0,1,2,3,4,5,6,7,8,9,10,11,12,13,14,15);
}

// ---------------- generic bf16 WMMA GEMM:  C[M,N] = A[M,K] * W[N,K]^T (+epilogue) ----
// EPI: 0 = none, 1 = softplus(acc + bias[n]), 2 = acc + resid[m,n]
// block = 256 threads (8 waves); block tile = 128(M) x WN*16(N).
// Weight tile (WN*16 x 32) is staged in LDS via CDNA5 async global->LDS copies,
// double-buffered so the fill overlaps the WMMAs; ASYNCcnt + barrier synchronize.
// Launch requirement: M % 128 == 0, N % (WN*16) == 0, K % 32 == 0 (exact tiling,
// so EXEC is all-ones around every WMMA and the inner loop is branch-free).
template<int WN, int EPI>
__global__ __launch_bounds__(256)
void gemm_wmma_kernel(const __bf16* __restrict__ A, const __bf16* __restrict__ W,
                      float* __restrict__ C, int M, int N, int K,
                      const float* __restrict__ bias, const float* __restrict__ resid) {
  constexpr int BROWS = WN * 16;
  __shared__ alignas(16) __bf16 smemB[2][BROWS][PADK];

  const int tid  = threadIdx.x;
  const int lane = tid & 31;
  const int wave = tid >> 5;
  const int row0 = (blockIdx.y * 8 + wave) * 16;
  const int col0 = blockIdx.x * BROWS;

  // async-copy one (BROWS x 32) weight tile into LDS buffer `buf`
  auto issue_copy = [&](int buf, int k0) {
    const int nchunks = BROWS * 4;                 // 16-byte chunks
    for (int i = tid; i < nchunks; i += 256) {
      int n = i >> 2, c = i & 3;
      const __bf16* g = W + (size_t)(col0 + n) * K + k0 + c * 8;
      unsigned lds = (unsigned)(uintptr_t)(&smemB[buf][n][c * 8]);
      asm volatile("global_load_async_to_lds_b128 %0, %1, off"
                   :: "v"(lds), "v"(g) : "memory");
    }
  };

  issue_copy(0, 0);
  asm volatile("s_wait_asynccnt 0" ::: "memory");
  __syncthreads();

  v8f acc[WN];
  #pragma unroll
  for (int j = 0; j < WN; ++j) acc[j] = v8f{0.f,0.f,0.f,0.f,0.f,0.f,0.f,0.f};

  int buf = 0;
  for (int k0 = 0; k0 < K; k0 += 32) {
    if (k0 + 32 < K) issue_copy(buf ^ 1, k0 + 32);   // prefetch next tile (async)
    v16bf a = load_A_frag(A, K, row0, k0, lane);
    #pragma unroll
    for (int j = 0; j < WN; ++j) {
      v16bf b = load_B_frag_lds(&smemB[buf][0][0], j, lane);
      acc[j] = __builtin_amdgcn_wmma_f32_16x16x32_bf16(false, a, false, b,
                                                       (short)0, acc[j], false, false);
    }
    asm volatile("s_wait_asynccnt 0" ::: "memory");  // next tile resident
    __syncthreads();                                  // all waves done reading `buf`
    buf ^= 1;
  }

  // C/D layout: VGPR r -> row (r + 8*(lane>=16)), col = lane&15
  const int mb = row0 + ((lane >> 4) << 3);
  const int nl = lane & 15;
  #pragma unroll
  for (int j = 0; j < WN; ++j) {
    int c = col0 + j * 16 + nl;
    #pragma unroll
    for (int r = 0; r < 8; ++r) {
      float v = acc[j][r];
      size_t idx = (size_t)(mb + r) * N + c;
      if (EPI == 1) { v += bias[c]; v = (v > 20.f) ? v : log1pf(__expf(v)); }
      else if (EPI == 2) { v += resid[idx]; }
      C[idx] = v;
    }
  }
}

// ---------------- weight conversion kernels ----------------
__global__ __launch_bounds__(256)
void f2bf_kernel(const float* __restrict__ src, __bf16* __restrict__ dst, int n) {
  int g = blockIdx.x * 256 + threadIdx.x;
  if (g < n) dst[g] = (__bf16)src[g];
}

__global__ __launch_bounds__(256)
void wdt_pad_kernel(const float* __restrict__ w, __bf16* __restrict__ dst) {
  int g = blockIdx.x * 256 + threadIdx.x;           // DINNER * DTPAD
  if (g >= DINNER * DTPAD) return;
  int k = g & (DTPAD - 1);
  int n = g >> 6;
  dst[g] = (k < DTRANK) ? (__bf16)w[n * DTRANK + k] : (__bf16)0.f;
}

__global__ __launch_bounds__(256)
void dt_pad_kernel(const float* __restrict__ dbc, __bf16* __restrict__ dst) {
  int g = blockIdx.x * 256 + threadIdx.x;           // NTOK * DTPAD
  if (g >= NTOK * DTPAD) return;
  int k = g & (DTPAD - 1);
  size_t tok = (size_t)(g >> 6);
  dst[g] = (k < DTRANK) ? (__bf16)dbc[tok * DBCN + k] : (__bf16)0.f;
}

// ---------------- LayerNorm -> bf16 ----------------
__global__ __launch_bounds__(256)
void ln_kernel(const float* __restrict__ x, const float* __restrict__ w,
               const float* __restrict__ b, __bf16* __restrict__ out) {
  const int tok = blockIdx.x;
  const float* p = x + (size_t)tok * DMODEL;
  float s = 0.f, s2 = 0.f;
  for (int i = threadIdx.x; i < DMODEL; i += 256) { float v = p[i]; s += v; s2 += v * v; }
  #pragma unroll
  for (int o = 16; o > 0; o >>= 1) { s += __shfl_down(s, o); s2 += __shfl_down(s2, o); }
  __shared__ float red[16];
  const int wave = threadIdx.x >> 5, lane = threadIdx.x & 31;
  if (lane == 0) { red[wave] = s; red[8 + wave] = s2; }
  __syncthreads();
  if (threadIdx.x == 0) {
    float S = 0.f, S2 = 0.f;
    #pragma unroll
    for (int i = 0; i < 8; ++i) { S += red[i]; S2 += red[8 + i]; }
    red[0] = S; red[8] = S2;
  }
  __syncthreads();
  const float mu  = red[0] * (1.f / DMODEL);
  const float var = red[8] * (1.f / DMODEL) - mu * mu;
  const float inv = rsqrtf(var + 1e-5f);
  __bf16* o = out + (size_t)tok * DMODEL;
  for (int i = threadIdx.x; i < DMODEL; i += 256)
    o[i] = (__bf16)((p[i] - mu) * inv * w[i] + b[i]);
}

// ---------------- depthwise causal conv + SiLU ----------------
__global__ __launch_bounds__(256)
void conv_silu_kernel(const float* __restrict__ xz, const float* __restrict__ cw,
                      const float* __restrict__ cb, float* __restrict__ u,
                      __bf16* __restrict__ u_bf) {
  size_t g = (size_t)blockIdx.x * 256 + threadIdx.x;  // BATCH*LSEQ*DINNER
  if (g >= (size_t)NTOK * DINNER) return;
  int d = (int)(g % DINNER);
  size_t bt = g / DINNER;
  int t = (int)(bt % LSEQ);
  int b = (int)(bt / LSEQ);
  float acc = cb[d];
  #pragma unroll
  for (int j = 0; j < DCONV; ++j) {
    int tt = t - (DCONV - 1) + j;
    if (tt >= 0)
      acc += xz[((size_t)b * LSEQ + tt) * XZDIM + d] * cw[d * DCONV + j];
  }
  float sv = acc / (1.f + __expf(-acc));
  u[g] = sv;
  u_bf[g] = (__bf16)sv;
}

// ---------------- fused selective scan + gating -> y_bf16 ----------------
// one lane per (b, d, s); 16-lane shfl_xor reduction over s; lane s==0 writes
__global__ __launch_bounds__(256)
void scan_kernel(const float* __restrict__ delta, const float* __restrict__ u,
                 const float* __restrict__ dbc,  const float* __restrict__ xz,
                 const float* __restrict__ A_log, const float* __restrict__ D_ssm,
                 __bf16* __restrict__ y_bf) {
  int g = blockIdx.x * 256 + threadIdx.x;       // BATCH*DINNER*DSTATE = 98304
  if (g >= BATCH * DINNER * DSTATE) return;
  const int s   = g & 15;
  const int g16 = g >> 4;
  const int d   = g16 % DINNER;
  const int b   = g16 / DINNER;

  const float Aval = -__expf(A_log[d * DSTATE + s]);
  const float Dval = D_ssm[d];

  const float* dptr  = delta + (size_t)b * LSEQ * DINNER + d;
  const float* uptr  = u     + (size_t)b * LSEQ * DINNER + d;
  const float* zptr  = xz    + (size_t)b * LSEQ * XZDIM + DINNER + d;
  const float* bcptr = dbc   + (size_t)b * LSEQ * DBCN + DTRANK + s;
  __bf16*      yptr  = y_bf  + (size_t)b * LSEQ * DINNER + d;

  float h = 0.f;
  for (int t = 0; t < LSEQ; ++t) {
    if (t + 8 < LSEQ) {   // cover the serial dependent-load chain
      __builtin_prefetch(dptr  + (size_t)(t + 8) * DINNER, 0, 1);
      __builtin_prefetch(uptr  + (size_t)(t + 8) * DINNER, 0, 1);
      __builtin_prefetch(bcptr + (size_t)(t + 8) * DBCN,   0, 1);
      __builtin_prefetch(zptr  + (size_t)(t + 8) * XZDIM,  0, 1);
    }
    float dt = dptr[(size_t)t * DINNER];
    float ut = uptr[(size_t)t * DINNER];
    float Bt = bcptr[(size_t)t * DBCN];
    float Ct = bcptr[(size_t)t * DBCN + DSTATE];
    float dA = __expf(dt * Aval);
    h = dA * h + (dt * ut) * Bt;
    float p = h * Ct;
    p += __shfl_xor(p, 1);
    p += __shfl_xor(p, 2);
    p += __shfl_xor(p, 4);
    p += __shfl_xor(p, 8);
    if (s == 0) {
      float yv = p + ut * Dval;
      float z  = zptr[(size_t)t * XZDIM];
      yv *= z / (1.f + __expf(-z));
      yptr[(size_t)t * DINNER] = (__bf16)yv;
    }
  }
}

// ---------------- host-side launch ----------------
extern "C" void kernel_launch(void* const* d_in, const int* in_sizes, int n_in,
                              void* d_out, int out_size, void* d_ws, size_t ws_size,
                              hipStream_t stream) {
  const float* input  = (const float*)d_in[0];
  const float* ln_w   = (const float*)d_in[1];
  const float* ln_b   = (const float*)d_in[2];
  const float* W_in   = (const float*)d_in[3];
  const float* conv_w = (const float*)d_in[4];
  const float* conv_b = (const float*)d_in[5];
  const float* W_x    = (const float*)d_in[6];
  const float* W_dt   = (const float*)d_in[7];
  const float* b_dt   = (const float*)d_in[8];
  const float* A_log  = (const float*)d_in[9];
  const float* D_ssm  = (const float*)d_in[10];
  const float* W_out  = (const float*)d_in[11];
  float* out = (float*)d_out;

  // workspace carve-up (256-byte aligned)
  char* base = (char*)d_ws;
  size_t off = 0;
  auto carve = [&](size_t bytes) -> char* {
    char* p = base + off;
    off = (off + bytes + 255) & ~(size_t)255;
    return p;
  };
  __bf16* w_in_bf  = (__bf16*)carve((size_t)XZDIM * DMODEL * 2);
  __bf16* w_x_bf   = (__bf16*)carve((size_t)DBCN * DINNER * 2);
  __bf16* w_dt_bf  = (__bf16*)carve((size_t)DINNER * DTPAD * 2);
  __bf16* w_out_bf = (__bf16*)carve((size_t)DMODEL * DINNER * 2);
  __bf16* xn_bf    = (__bf16*)carve((size_t)NTOK * DMODEL * 2);
  float*  xz       = (float*) carve((size_t)NTOK * XZDIM * 4);
  float*  u        = (float*) carve((size_t)NTOK * DINNER * 4);
  __bf16* u_bf     = (__bf16*)carve((size_t)NTOK * DINNER * 2);
  float*  dbc      = (float*) carve((size_t)NTOK * DBCN * 4);
  __bf16* dt_bf    = (__bf16*)carve((size_t)NTOK * DTPAD * 2);
  float*  delta    = (float*) carve((size_t)NTOK * DINNER * 4);
  __bf16* y_bf     = (__bf16*)carve((size_t)NTOK * DINNER * 2);

  dim3 blk(256);
  auto cdiv = [](int a, int b) { return (a + b - 1) / b; };

  // 1) weights -> bf16
  f2bf_kernel<<<cdiv(XZDIM * DMODEL, 256), blk, 0, stream>>>(W_in, w_in_bf, XZDIM * DMODEL);
  f2bf_kernel<<<cdiv(DBCN * DINNER, 256), blk, 0, stream>>>(W_x, w_x_bf, DBCN * DINNER);
  wdt_pad_kernel<<<cdiv(DINNER * DTPAD, 256), blk, 0, stream>>>(W_dt, w_dt_bf);
  f2bf_kernel<<<cdiv(DMODEL * DINNER, 256), blk, 0, stream>>>(W_out, w_out_bf, DMODEL * DINNER);

  // 2) LayerNorm -> bf16
  ln_kernel<<<NTOK, blk, 0, stream>>>(input, ln_w, ln_b, xn_bf);

  // 3) xz = xn @ W_in^T     [8192,768] x [3072,768]^T   (N=3072: WN=4, gx=48)
  gemm_wmma_kernel<4,0><<<dim3(XZDIM / 64, NTOK / 128), blk, 0, stream>>>(
      xn_bf, w_in_bf, xz, NTOK, XZDIM, DMODEL, nullptr, nullptr);

  // 4) depthwise causal conv + SiLU -> u (f32 + bf16)
  conv_silu_kernel<<<cdiv(NTOK * DINNER, 256), blk, 0, stream>>>(xz, conv_w, conv_b, u, u_bf);

  // 5) dbc = u @ W_x^T      [8192,1536] x [80,1536]^T   (N=80: WN=5, gx=1)
  gemm_wmma_kernel<5,0><<<dim3(1, NTOK / 128), blk, 0, stream>>>(
      u_bf, w_x_bf, dbc, NTOK, DBCN, DINNER, nullptr, nullptr);

  // 6) dt slice -> bf16 padded K=64
  dt_pad_kernel<<<cdiv(NTOK * DTPAD, 256), blk, 0, stream>>>(dbc, dt_bf);

  // 7) delta = softplus(dt @ W_dt^T + b_dt)   [8192,64] x [1536,64]^T  (WN=4, gx=24)
  gemm_wmma_kernel<4,1><<<dim3(DINNER / 64, NTOK / 128), blk, 0, stream>>>(
      dt_bf, w_dt_bf, delta, NTOK, DINNER, DTPAD, b_dt, nullptr);

  // 8) fused selective scan + u*D + silu(z) gate -> y_bf
  scan_kernel<<<cdiv(BATCH * DINNER * DSTATE, 256), blk, 0, stream>>>(
      delta, u, dbc, xz, A_log, D_ssm, y_bf);

  // 9) out = y @ W_out^T + skip   [8192,1536] x [768,1536]^T  (WN=4, gx=12)
  gemm_wmma_kernel<4,2><<<dim3(DMODEL / 64, NTOK / 128), blk, 0, stream>>>(
      y_bf, w_out_bf, out, NTOK, DMODEL, DINNER, nullptr, input);
}
